// FastOLy_56169582297263
// MI455X (gfx1250) — matrix operand order repaired and checked by reading, not compiled, not run on hardware
//
#include <hip/hip_runtime.h>
#include <hip/hip_bf16.h>
#include <math.h>

// ---------------------------------------------------------------------------
// CDNA5 (gfx1250) wave32 transformer forward.
//  - All GEMMs + attention matmuls: v_wmma_f32_16x16x32_bf16.
//  - A-side tile staging via Tensor Data Mover (tensor_load_to_lds, TENSORcnt)
//  - B-side f32 weights converted to bf16 in a register-pipelined double
//    buffer (L2 absorbs re-reads; 131MB head weight < 192MB L2).
//  - qkv activations kept bf16 end-to-end.
// ---------------------------------------------------------------------------

typedef __attribute__((ext_vector_type(16))) __bf16 v16bf;
typedef __attribute__((ext_vector_type(8)))  float  v8f;
typedef __attribute__((ext_vector_type(4)))  unsigned int u32x4;
typedef __attribute__((ext_vector_type(8)))  int i32x8;
typedef __attribute__((ext_vector_type(4)))  int i32x4;

union FragU { uint4 u[2]; v16bf v; };
union BU    { uint4 u; __bf16 h[8]; };

// LDS byte offset of a __shared__ object (generic -> AS(3) -> integer).
__device__ __forceinline__ unsigned lds_addr_of(const void* p) {
  return (unsigned)(unsigned long long)(const __attribute__((address_space(3))) void*)p;
}

// Issue one TDM 2D tile load: tile_h rows x tile_w bf16 elems, row stride
// `stride` elems in global, packed contiguously into LDS at `lds`.
// D# group0: count=1, lds_addr, 57b global_addr, type=2 ("image").
// D# group1: data_size=2B, tensor dims == tile dims (tiles always in-bounds),
//            tensor_dim0_stride = stride. Groups 2/3 zero (2D tensor).
__device__ __forceinline__ void tdm_load_2d(unsigned lds, unsigned long long ga,
                                            int tile_w, int tile_h, int stride) {
  u32x4 g0;
  g0[0] = 1u;                                              // count = 1
  g0[1] = lds;                                             // lds_addr
  g0[2] = (unsigned)ga;                                    // global_addr[31:0]
  g0[3] = (unsigned)((ga >> 32) & 0x1FFFFFFull) | (2u << 30); // addr[56:32]|type=2
  i32x8 g1;
  g1[0] = (int)(1u << 16);                                 // data_size = 1 (2B)
  g1[1] = (int)(((unsigned)tile_w & 0xFFFFu) << 16);       // tensor_dim0 lo
  g1[2] = (int)((((unsigned)tile_w) >> 16) |
                (((unsigned)tile_h & 0xFFFFu) << 16));     // dim0 hi | dim1 lo
  g1[3] = (int)((((unsigned)tile_h) >> 16) |
                (((unsigned)tile_w) << 16));               // dim1 hi | tile_dim0
  g1[4] = (int)((unsigned)tile_h & 0xFFFFu);               // tile_dim1 (tile_dim2=0)
  g1[5] = stride;                                          // tensor_dim0_stride lo
  g1[6] = 0;                                               // stride hi | dim1_stride
  g1[7] = 0;
  i32x4 z4 = {0, 0, 0, 0};
  i32x8 z8 = {0, 0, 0, 0, 0, 0, 0, 0};
  // clang-23 / amdgpu-toolchain 6-arg form: (g0, g1, g2, g3, g4, cpol)
  __builtin_amdgcn_tensor_load_to_lds(g0, g1, z4, z4, z8, 0);
}

// A fragment (16x32 bf16, row-major LDS, row stride = `stride` elems).
// lanes 0-15 -> M=lane, K = {0..15}; lanes 16-31 -> M=lane-16, K={16..31}
__device__ __forceinline__ v16bf load_a_frag(const __bf16* base, int stride,
                                             int hl, int m) {
  const __bf16* row = base + m * stride;
  FragU f;
  f.u[0] = *(const uint4*)(row + hl * 16);
  f.u[1] = *(const uint4*)(row + hl * 16 + 8);
  return f.v;
}

// B fragment (32x16). LDS holds B as [n][k] so each lane's 16 K are contiguous.
__device__ __forceinline__ v16bf load_b_frag(const __bf16* base, int stride,
                                             int hl, int n) {
  const __bf16* row = base + n * stride + hl * 16;
  FragU f;
  f.u[0] = *(const uint4*)(row);
  f.u[1] = *(const uint4*)(row + 8);
  return f.v;
}

__device__ __forceinline__ v8f wmma_bf16(v16bf a, v16bf b, v8f c) {
  return __builtin_amdgcn_wmma_f32_16x16x32_bf16(
      false, a, false, b, (short)0, c, false, false);
}

// ---------------------------------------------------------------------------
// Kernel 1: first-response position per batch row.
// ---------------------------------------------------------------------------
__global__ void resp_kernel(const int* __restrict__ idx, int* __restrict__ resp) {
  __shared__ int smin;
  if (threadIdx.x == 0) smin = 1024;
  __syncthreads();
  if (idx[blockIdx.x * 1024 + threadIdx.x] == 2) atomicMin(&smin, (int)threadIdx.x);
  __syncthreads();
  if (threadIdx.x == 0) resp[blockIdx.x] = smin;
}

// ---------------------------------------------------------------------------
// Kernel 2: token emb gather + pos emb + dialog emb -> x (f32)
// ---------------------------------------------------------------------------
__global__ __launch_bounds__(256) void embed_kernel(
    const int* __restrict__ idx, const float* __restrict__ tok,
    const float* __restrict__ qe, const float* __restrict__ ae,
    const float* __restrict__ pos, const int* __restrict__ resp,
    float* __restrict__ x) {
  int row = blockIdx.x;
  int b = row >> 10, t = row & 1023;
  int c = threadIdx.x * 4;
  const float* te = tok + (size_t)idx[row] * 1024;
  const float* de = (t >= resp[b]) ? ae : qe;
  float4 v = *(const float4*)(te + c);
  float4 p = *(const float4*)(pos + (size_t)t * 1024 + c);
  float4 d = *(const float4*)(de + c);
  float4 o;
  o.x = v.x + p.x + d.x; o.y = v.y + p.y + d.y;
  o.z = v.z + p.z + d.z; o.w = v.w + p.w + d.w;
  *(float4*)(x + (size_t)row * 1024 + c) = o;
}

// ---------------------------------------------------------------------------
// Kernel 3: LayerNorm (f32) -> bf16. One block per row, C=1024.
// ---------------------------------------------------------------------------
__global__ __launch_bounds__(256) void ln_bf16_kernel(
    const float* __restrict__ x, const float* __restrict__ s,
    const float* __restrict__ b, __bf16* __restrict__ out) {
  const int C = 1024;
  int row = blockIdx.x, t = threadIdx.x;
  float4 v = *(const float4*)(x + (size_t)row * C + t * 4);
  float sum = v.x + v.y + v.z + v.w;
  float sq  = v.x * v.x + v.y * v.y + v.z * v.z + v.w * v.w;
#pragma unroll
  for (int off = 16; off > 0; off >>= 1) {
    sum += __shfl_down(sum, off, 32);
    sq  += __shfl_down(sq,  off, 32);
  }
  __shared__ float rs[8], rq[8], mv[2];
  int wid = t >> 5, lane = t & 31;
  if (lane == 0) { rs[wid] = sum; rq[wid] = sq; }
  __syncthreads();
  if (t == 0) {
    float ts = 0.f, tq = 0.f;
#pragma unroll
    for (int i = 0; i < 8; ++i) { ts += rs[i]; tq += rq[i]; }
    float mean = ts * (1.f / 1024.f);
    float var  = tq * (1.f / 1024.f) - mean * mean;
    mv[0] = mean; mv[1] = rsqrtf(var + 1e-5f);
  }
  __syncthreads();
  float mean = mv[0], rstd = mv[1];
  float xv[4] = {v.x, v.y, v.z, v.w};
#pragma unroll
  for (int i = 0; i < 4; ++i) {
    int c = t * 4 + i;
    out[(size_t)row * C + c] = (__bf16)((xv[i] - mean) * rstd * s[c] + b[c]);
  }
}

// ---------------------------------------------------------------------------
// B-tile register pipeline helpers (f32 -> bf16, LDS layout [n][k], BK=32).
// ---------------------------------------------------------------------------
template <bool TRANSB>
__device__ __forceinline__ void load_B_regs(const float* __restrict__ B, int k0,
                                            int n0, int N, int K, int tid,
                                            float4 (&r4)[4]) {
#pragma unroll
  for (int it = 0; it < 4; ++it) {
    int i4 = tid + it * 256;
    if (TRANSB) {
      int nr = i4 >> 3, c = (i4 & 7) * 4;
      r4[it] = *(const float4*)(B + (size_t)(n0 + nr) * K + k0 + c);
    } else {
      int kr = i4 >> 5, c = i4 & 31;
      r4[it] = *(const float4*)(B + (size_t)(k0 + kr) * N + n0 + c * 4);
    }
  }
}

template <bool TRANSB>
__device__ __forceinline__ void store_B_lds(__bf16* __restrict__ Bls, int tid,
                                            const float4 (&r4)[4]) {
#pragma unroll
  for (int it = 0; it < 4; ++it) {
    int i4 = tid + it * 256;
    float4 f = r4[it];
    if (TRANSB) {
      int nr = i4 >> 3, c = (i4 & 7) * 4;
      __bf16* d = &Bls[nr * 32 + c];
      d[0] = (__bf16)f.x; d[1] = (__bf16)f.y;
      d[2] = (__bf16)f.z; d[3] = (__bf16)f.w;
    } else {
      int kr = i4 >> 5, c = i4 & 31;
      int n = c * 4;
      Bls[(n + 0) * 32 + kr] = (__bf16)f.x;
      Bls[(n + 1) * 32 + kr] = (__bf16)f.y;
      Bls[(n + 2) * 32 + kr] = (__bf16)f.z;
      Bls[(n + 3) * 32 + kr] = (__bf16)f.w;
    }
  }
}

// ---------------------------------------------------------------------------
// Kernel 4: bf16-WMMA GEMM, TDM-staged A, double-buffered LDS.
// EPI: 0 f32 out; 1 resid+out f32; 2 gelu(out+bias) bf16; 3 resid+out+bias f32;
//      4 plain bf16 out.
// TRANSB: false -> B [K,N]; true -> B [N,K] (tied head).
// Block tile 128x128x32, 256 threads (8 waves, each 64x32).
// ---------------------------------------------------------------------------
template <int EPI, bool TRANSB>
__global__ __launch_bounds__(256) void gemm_bf16_kernel(
    const __bf16* __restrict__ A, const float* __restrict__ B,
    float* __restrict__ Cf, __bf16* __restrict__ Cb,
    const float* __restrict__ bias, const float* __restrict__ resid,
    int M, int N, int K) {
  constexpr int BM = 128, BN = 128, BK = 32;
  __shared__ __align__(16) __bf16 Als[2][BM * BK];
  __shared__ __align__(16) __bf16 Bls[2][BN * BK];   // [n][k]

  int tid = threadIdx.x;
  int wid = tid >> 5, lane = tid & 31;
  int hl = lane >> 4, nl = lane & 15;
  int wr = wid >> 2, wc = wid & 3;                   // 2 x 4 wave grid
  int m0 = blockIdx.y * BM;
  int n0 = blockIdx.x * BN;

  v8f z = {};
  v8f acc[4][2];
#pragma unroll
  for (int i = 0; i < 4; ++i)
#pragma unroll
    for (int j = 0; j < 2; ++j) acc[i][j] = z;

  // ---- prologue: stage tile 0 (A via TDM, B via regs+cvt) ----
  if (tid < 32)
    tdm_load_2d(lds_addr_of(&Als[0][0]),
                (unsigned long long)(const void*)(A + (size_t)m0 * K),
                BK, BM, K);
  {
    float4 r0[4];
    load_B_regs<TRANSB>(B, 0, n0, N, K, tid, r0);
    store_B_lds<TRANSB>(&Bls[0][0], tid, r0);
  }
  if (tid < 32) __builtin_amdgcn_s_wait_tensorcnt(0);
  __syncthreads();

  int buf = 0;
  for (int k0 = 0; k0 < K; k0 += BK) {
    int nxt = buf ^ 1;
    bool more = (k0 + BK) < K;
    // DMA next A tile while this one computes
    if (more && tid < 32)
      tdm_load_2d(lds_addr_of(&Als[nxt][0]),
                  (unsigned long long)(const void*)(A + (size_t)m0 * K + k0 + BK),
                  BK, BM, K);
    float4 r4[4];
    if (more) load_B_regs<TRANSB>(B, k0 + BK, n0, N, K, tid, r4);
    if (k0 + 2 * BK < K) {    // hint tile after next (global_prefetch_b8)
      if (TRANSB)
        __builtin_prefetch(B + (size_t)(n0 + (tid & 127)) * K + k0 + 2 * BK, 0, 1);
      else
        __builtin_prefetch(B + (size_t)(k0 + 2 * BK + (tid & 31)) * N + n0, 0, 1);
    }

    v16bf af[4], bfg[2];
#pragma unroll
    for (int mb = 0; mb < 4; ++mb)
      af[mb] = load_a_frag(&Als[buf][(wr * 64 + mb * 16) * BK], BK, hl, nl);
#pragma unroll
    for (int nb = 0; nb < 2; ++nb)
      bfg[nb] = load_b_frag(&Bls[buf][(wc * 32 + nb * 16) * BK], BK, hl, nl);
#pragma unroll
    for (int mb = 0; mb < 4; ++mb)
#pragma unroll
      for (int nb = 0; nb < 2; ++nb)
        acc[mb][nb] = wmma_bf16(af[mb], bfg[nb], acc[mb][nb]);

    if (more) store_B_lds<TRANSB>(&Bls[nxt][0], tid, r4);
    if (tid < 32) __builtin_amdgcn_s_wait_tensorcnt(0);
    __syncthreads();
    buf = nxt;
  }

  // ---- epilogue ----
#pragma unroll
  for (int mb = 0; mb < 4; ++mb) {
#pragma unroll
    for (int nb = 0; nb < 2; ++nb) {
      int col = n0 + wc * 32 + nb * 16 + nl;
#pragma unroll
      for (int r = 0; r < 8; ++r) {
        int row = m0 + wr * 64 + mb * 16 + r + hl * 8;
        size_t o = (size_t)row * N + col;
        float v = acc[mb][nb][r];
        if (EPI == 0) {
          Cf[o] = v;
        } else if (EPI == 1) {
          Cf[o] = resid[o] + v;
        } else if (EPI == 2) {
          float tt = v + bias[col];
          Cb[o] = (__bf16)(0.5f * tt * (1.f + erff(tt * 0.70710678118f)));
        } else if (EPI == 3) {
          Cf[o] = resid[o] + v + bias[col];
        } else {
          Cb[o] = (__bf16)v;
        }
      }
    }
  }
}

// ---------------------------------------------------------------------------
// Kernel 5: flash attention (bf16 qkv). Grid (B*H, T/128); 8 waves; each wave
// owns 16 q rows; 32-key tiles; Q/K tiles via TDM; online softmax; WMMA both.
// ---------------------------------------------------------------------------
__global__ __launch_bounds__(256) void attn_kernel(
    const __bf16* __restrict__ qkv, __bf16* __restrict__ y) {
  const int C = 1024, D = 64, T = 1024, TC = 3072;
  __shared__ __align__(16) __bf16 Qls[128 * 64];
  __shared__ __align__(16) __bf16 Kls[32 * 64];    // [key][d]
  __shared__ __align__(16) __bf16 Vt[64 * 32];     // [d][key]
  __shared__ __align__(16) float  Sls[128 * 32];
  __shared__ __align__(16) __bf16 Pls[128 * 32];
  __shared__ float alpha_ls[128];
  __shared__ float l_ls[128];

  int tid = threadIdx.x, wid = tid >> 5, lane = tid & 31;
  int hl = lane >> 4, nl = lane & 15;
  int bh = blockIdx.x;
  int b = bh >> 4, h = bh & 15;
  int q0 = blockIdx.y * 128;
  size_t base = (size_t)b * T * TC + (size_t)h * D;

  // stage Q tile via TDM (128 rows x 64 bf16, row stride 3C)
  if (tid < 32)
    tdm_load_2d(lds_addr_of(Qls),
                (unsigned long long)(const void*)(qkv + base + (size_t)q0 * TC),
                64, 128, TC);
  float m_reg = -1e30f, l_reg = 0.f;
  v8f z = {};
  v8f o_acc[4] = {z, z, z, z};
  if (tid < 32) __builtin_amdgcn_s_wait_tensorcnt(0);
  __syncthreads();

  v16bf qf[2];
  qf[0] = load_a_frag(&Qls[(wid * 16) * 64 + 0], 64, hl, nl);
  qf[1] = load_a_frag(&Qls[(wid * 16) * 64 + 32], 64, hl, nl);

  const float scale = 0.125f;                      // 1/sqrt(64)
  int jend = q0 / 32 + 4;                          // causal bound
  for (int j = 0; j < jend; ++j) {
    int s0 = j * 32;
    __syncthreads();                               // protect Kls/Vt reuse
    if (tid < 32)                                  // K tile via TDM
      tdm_load_2d(lds_addr_of(Kls),
                  (unsigned long long)(const void*)(qkv + base + C + (size_t)s0 * TC),
                  64, 32, TC);
    {                                              // V tile, transposed [d][key]
      int r = tid >> 3, c = (tid & 7) * 8;
      BU bu;
      bu.u = *(const uint4*)(qkv + base + 2 * C + (size_t)(s0 + r) * TC + c);
#pragma unroll
      for (int i = 0; i < 8; ++i) Vt[(c + i) * 32 + r] = bu.h[i];
    }
    if (tid < 32) __builtin_amdgcn_s_wait_tensorcnt(0);
    __syncthreads();

    // S = Q @ K^T (16 rows x 32 keys per wave)
#pragma unroll
    for (int nb = 0; nb < 2; ++nb) {
      v8f sa = z;
      v16bf k0f = load_b_frag(&Kls[(nb * 16) * 64 + 0], 64, hl, nl);
      v16bf k1f = load_b_frag(&Kls[(nb * 16) * 64 + 32], 64, hl, nl);
      sa = wmma_bf16(qf[0], k0f, sa);
      sa = wmma_bf16(qf[1], k1f, sa);
#pragma unroll
      for (int r = 0; r < 8; ++r) {
        int rl = wid * 16 + r + hl * 8;
        float v = sa[r] * scale;
        if (s0 + nb * 16 + nl > q0 + rl) v = -1e30f;   // causal mask
        Sls[rl * 32 + nb * 16 + nl] = v;
      }
    }
    // online softmax: lanes 0-15 each own one row of this wave's 16
    if (lane < 16) {
      int rl = wid * 16 + lane;
      float mx = -1e30f;
#pragma unroll
      for (int c = 0; c < 32; ++c) mx = fmaxf(mx, Sls[rl * 32 + c]);
      float nm = fmaxf(m_reg, mx);
      float al = __expf(m_reg - nm);
      float sum = 0.f;
#pragma unroll
      for (int c = 0; c < 32; ++c) {
        float p = __expf(Sls[rl * 32 + c] - nm);
        Pls[rl * 32 + c] = (__bf16)p;
        sum += p;
      }
      l_reg = l_reg * al + sum;
      m_reg = nm;
      alpha_ls[rl] = al;
    }
#pragma unroll
    for (int r = 0; r < 8; ++r) {                  // rescale O by per-row alpha
      float al = alpha_ls[wid * 16 + r + hl * 8];
#pragma unroll
      for (int db = 0; db < 4; ++db) o_acc[db][r] *= al;
    }
    v16bf pf = load_a_frag(&Pls[(wid * 16) * 32], 32, hl, nl);
#pragma unroll
    for (int db = 0; db < 4; ++db) {               // O += P @ V
      v16bf vf = load_b_frag(&Vt[(db * 16) * 32], 32, hl, nl);
      o_acc[db] = wmma_bf16(pf, vf, o_acc[db]);
    }
  }

  if (lane < 16) l_ls[wid * 16 + lane] = 1.0f / l_reg;
#pragma unroll
  for (int r = 0; r < 8; ++r) {
    int rl = wid * 16 + r + hl * 8;
    float li = l_ls[rl];
    size_t row = (size_t)b * T + q0 + rl;
#pragma unroll
    for (int db = 0; db < 4; ++db)
      y[row * C + h * D + db * 16 + nl] = (__bf16)(o_acc[db][r] * li);
  }
}

// ---------------------------------------------------------------------------
// Host driver
// ---------------------------------------------------------------------------
extern "C" void kernel_launch(void* const* d_in, const int* in_sizes, int n_in,
                              void* d_out, int out_size, void* d_ws, size_t ws_size,
                              hipStream_t stream) {
  (void)in_sizes; (void)n_in; (void)out_size; (void)ws_size;
  const int C = 1024, L = 8, Vv = 32000, M = 4096;

  const int*   idx     = (const int*)d_in[0];
  const float* tok     = (const float*)d_in[1];
  const float* q_emb   = (const float*)d_in[2];
  const float* a_emb   = (const float*)d_in[3];
  const float* pos_emb = (const float*)d_in[4];
  const float* ln1_s   = (const float*)d_in[5];
  const float* ln1_b   = (const float*)d_in[6];
  const float* attn_w  = (const float*)d_in[7];
  const float* proj_w  = (const float*)d_in[8];
  const float* ln2_s   = (const float*)d_in[9];
  const float* ln2_b   = (const float*)d_in[10];
  const float* mlp_w1  = (const float*)d_in[11];
  const float* mlp_b1  = (const float*)d_in[12];
  const float* mlp_w2  = (const float*)d_in[13];
  const float* mlp_b2  = (const float*)d_in[14];
  const float* lnf_s   = (const float*)d_in[15];
  const float* lnf_b   = (const float*)d_in[16];

  char* ws = (char*)d_ws;
  float*  x    = (float*)(ws + 0);            // 16 MB f32  [4096,1024]
  __bf16* qkv  = (__bf16*)(ws + 16777216);    // 24 MB bf16 [4096,3072]
  __bf16* hbuf = (__bf16*)(ws + 41943040);    //  8 MB bf16 [4096,1024]
  __bf16* h2   = (__bf16*)(ws + 50331648);    // 16 MB bf16 [4096,2048]
  __bf16* ybuf = (__bf16*)(ws + 67108864);    //  8 MB bf16 [4096,1024]
  int*    resp = (int*)(ws + 75497472);       // 16 B

  dim3 blk(256);

  resp_kernel<<<4, 1024, 0, stream>>>(idx, resp);
  embed_kernel<<<M, blk, 0, stream>>>(idx, tok, q_emb, a_emb, pos_emb, resp, x);

  for (int l = 0; l < L; ++l) {
    ln_bf16_kernel<<<M, blk, 0, stream>>>(x, ln1_s + (size_t)l * C,
                                          ln1_b + (size_t)l * C, hbuf);
    // qkv = h @ attn_w  -> bf16
    gemm_bf16_kernel<4, false><<<dim3(3072 / 128, M / 128), blk, 0, stream>>>(
        hbuf, attn_w + (size_t)l * C * 3 * C, nullptr, qkv, nullptr, nullptr,
        M, 3 * C, C);
    attn_kernel<<<dim3(64, 8), blk, 0, stream>>>(qkv, ybuf);
    // x = x + y @ proj_w
    gemm_bf16_kernel<1, false><<<dim3(C / 128, M / 128), blk, 0, stream>>>(
        ybuf, proj_w + (size_t)l * C * C, x, nullptr, nullptr, x, M, C, C);
    ln_bf16_kernel<<<M, blk, 0, stream>>>(x, ln2_s + (size_t)l * C,
                                          ln2_b + (size_t)l * C, hbuf);
    // h2 = gelu(h @ w1 + b1) -> bf16
    gemm_bf16_kernel<2, false><<<dim3(2 * C / 128, M / 128), blk, 0, stream>>>(
        hbuf, mlp_w1 + (size_t)l * C * 2 * C, nullptr, h2,
        mlp_b1 + (size_t)l * 2 * C, nullptr, M, 2 * C, C);
    // x = x + h2 @ w2 + b2
    gemm_bf16_kernel<3, false><<<dim3(C / 128, M / 128), blk, 0, stream>>>(
        h2, mlp_w2 + (size_t)l * 2 * C * C, x, nullptr,
        mlp_b2 + (size_t)l * C, x, M, C, 2 * C);
  }

  ln_bf16_kernel<<<M, blk, 0, stream>>>(x, lnf_s, lnf_b, hbuf);
  gemm_bf16_kernel<0, true><<<dim3(Vv / 128, M / 128), blk, 0, stream>>>(
      hbuf, tok, (float*)d_out, nullptr, nullptr, nullptr, M, Vv, C);
}
